// LSTM_20169166422058
// MI455X (gfx1250) — matrix operand order repaired and checked by reading, not compiled, not run on hardware
//
#include <hip/hip_runtime.h>
#include <hip/hip_bf16.h>
#include <math.h>

// ---------------------------------------------------------------------------
// NRNM memory-LSTM for MI455X (gfx1250, wave32, WMMA).
// GEMMs via v_wmma_f32_16x16x32_bf16, register-blocked 32Mx64N per wave
// (2 A-frags x 4 B-frags -> 8 WMMAs per k-step, 0.75 loads/WMMA).
// ---------------------------------------------------------------------------

typedef __attribute__((ext_vector_type(16))) __bf16 v16bf;
typedef __attribute__((ext_vector_type(8)))  float  v8f;

#define T_STEPS 128
#define BATCH   256
#define DIN0    256
#define HID     512
#define BH      (BATCH * HID)          // 131072
#define N4H     (4 * HID)              // 2048
#define MSIZE   256
#define SLOTS   8
#define INV_SQRT_M 0.0625f

// ---------------------------------------------------------------------------
// Pack fp32 activations (row-major M x K, up to 3 concatenated segments along
// K) into WMMA A-fragment layout (16-bit A 16x32, per CDNA5 ISA 7.12.2):
//   lanes 0-15 : M = mt*16+lane,    K = kt*32 + (j<8 ? j : j+8)
//   lanes 16-31: M = mt*16+lane-16, K = kt*32 + 8 + (j<8 ? j : j+8)
// storage: dst[(((mt*KT)+kt)*32 + lane)*16 + j]
// ---------------------------------------------------------------------------
__global__ __launch_bounds__(256) void pack_a_kernel(
    const float* __restrict__ s0, int w0,
    const float* __restrict__ s1, int w1,
    const float* __restrict__ s2, int w2,
    __bf16* __restrict__ dst, int KT) {
  size_t id = (size_t)blockIdx.x * blockDim.x + threadIdx.x;
  size_t total = (size_t)BATCH * KT * 32u;   // halves total = M*K
  if (id >= total) return;
  int j    = (int)(id & 15);
  int lane = (int)((id >> 4) & 31);
  size_t rest = id >> 9;
  int kt = (int)(rest % (size_t)KT);
  int mt = (int)(rest / (size_t)KT);
  int m  = mt * 16 + (lane & 15);
  int k  = kt * 32 + ((lane >= 16) ? 8 : 0) + (j < 8 ? j : j + 8);
  float v;
  if (k < w0)            v = s0[(size_t)m * w0 + k];
  else if (k < w0 + w1)  v = s1[(size_t)m * w1 + (k - w0)];
  else                   v = s2[(size_t)m * w2 + (k - w0 - w1)];
  dst[id] = (__bf16)v;
}

// ---------------------------------------------------------------------------
// Pack fp32 weights (row-major K x N, up to 3 segments stacked along K) into
// WMMA B-fragment layout:
//   lanes 0-15 : N = nt*16+lane,    K = kt*32 + (j<8 ? j : j+8)
//   lanes 16-31: N = nt*16+lane-16, K = kt*32 + 8 + (j<8 ? j : j+8)
// storage: dst[(((nt*KT)+kt)*32 + lane)*16 + j]
// ---------------------------------------------------------------------------
__global__ __launch_bounds__(256) void pack_b_kernel(
    const float* __restrict__ s0, int k0,
    const float* __restrict__ s1, int k1,
    const float* __restrict__ s2, int k2,
    int N, __bf16* __restrict__ dst, int KT) {
  size_t id = (size_t)blockIdx.x * blockDim.x + threadIdx.x;
  size_t total = (size_t)N * KT * 32u;       // halves total = K*N
  if (id >= total) return;
  int j    = (int)(id & 15);
  int lane = (int)((id >> 4) & 31);
  size_t rest = id >> 9;
  int kt = (int)(rest % (size_t)KT);
  int nt = (int)(rest / (size_t)KT);
  int n  = nt * 16 + (lane & 15);
  int k  = kt * 32 + ((lane >= 16) ? 8 : 0) + (j < 8 ? j : j + 8);
  float v;
  if (k < k0)            v = s0[(size_t)k * N + n];
  else if (k < k0 + k1)  v = s1[(size_t)(k - k0) * N + n];
  else                   v = s2[(size_t)(k - k0 - k1) * N + n];
  dst[id] = (__bf16)v;
}

// ---------------------------------------------------------------------------
// WMMA GEMM, register-blocked: each wave computes a 32(M) x 64(N) block of C
// as a 2x4 grid of 16x16 WMMA tiles.  Per k-step: 2 A-frag + 4 B-frag loads
// feed 8 v_wmma_f32_16x16x32_bf16 (8 independent accumulators for ILP).
// C = A*B [+ bias[n]] [tanh].  Requires M%32==0, N%64==0 (always true here).
// ---------------------------------------------------------------------------
__global__ __launch_bounds__(256) void wmma_gemm_kernel(
    const __bf16* __restrict__ Ap, const __bf16* __restrict__ Bp,
    const float* __restrict__ bias, float* __restrict__ C,
    int Nb /* N/64 */, int KT /* K/32 */, int N, int Mb /* M/32 */,
    int ep_tanh) {
  int wave = threadIdx.x >> 5;
  int lane = threadIdx.x & 31;
  int blk  = blockIdx.x * 8 + wave;
  if (blk >= Mb * Nb) return;                // wave-uniform guard
  int mb = blk / Nb;
  int nb = blk % Nb;

  const size_t tstride = (size_t)KT * 32;    // v16bf per 16-wide tile column
  const v16bf* __restrict__ Af = (const v16bf*)Ap + (size_t)(2 * mb) * tstride + lane;
  const v16bf* __restrict__ Bf = (const v16bf*)Bp + (size_t)(4 * nb) * tstride + lane;

  v8f acc[8];
#pragma unroll
  for (int i = 0; i < 8; ++i) acc[i] = (v8f){};

  for (int kt = 0; kt < KT; ++kt) {
    const size_t ko = (size_t)kt * 32;
    v16bf a0 = Af[ko];
    v16bf a1 = Af[tstride + ko];
    v16bf b0 = Bf[ko];
    v16bf b1 = Bf[tstride + ko];
    v16bf b2 = Bf[2 * tstride + ko];
    v16bf b3 = Bf[3 * tstride + ko];
    acc[0] = __builtin_amdgcn_wmma_f32_16x16x32_bf16(false, a0, false, b0, (short)0, acc[0], false, false);
    acc[1] = __builtin_amdgcn_wmma_f32_16x16x32_bf16(false, a0, false, b1, (short)0, acc[1], false, false);
    acc[2] = __builtin_amdgcn_wmma_f32_16x16x32_bf16(false, a0, false, b2, (short)0, acc[2], false, false);
    acc[3] = __builtin_amdgcn_wmma_f32_16x16x32_bf16(false, a0, false, b3, (short)0, acc[3], false, false);
    acc[4] = __builtin_amdgcn_wmma_f32_16x16x32_bf16(false, a1, false, b0, (short)0, acc[4], false, false);
    acc[5] = __builtin_amdgcn_wmma_f32_16x16x32_bf16(false, a1, false, b1, (short)0, acc[5], false, false);
    acc[6] = __builtin_amdgcn_wmma_f32_16x16x32_bf16(false, a1, false, b2, (short)0, acc[6], false, false);
    acc[7] = __builtin_amdgcn_wmma_f32_16x16x32_bf16(false, a1, false, b3, (short)0, acc[7], false, false);
  }

  // C layout (ISA 7.12.2): VGPR r, lanes 0-15 -> M = mt*16 + r, N = nt*16+lane
  //                                 lanes 16-31 -> M = mt*16 + 8 + r
  int lh = lane & 15;
  int hb = (lane >> 4) << 3;
#pragma unroll
  for (int mi = 0; mi < 2; ++mi) {
#pragma unroll
    for (int ni = 0; ni < 4; ++ni) {
      v8f a = acc[mi * 4 + ni];
      int n  = (4 * nb + ni) * 16 + lh;
      int m0 = (2 * mb + mi) * 16 + hb;
      float bv = bias ? bias[n] : 0.0f;
#pragma unroll
      for (int r = 0; r < 8; ++r) {
        float v = a[r] + bv;
        if (ep_tanh) v = tanhf(v);
        C[(size_t)(m0 + r) * N + n] = v;
      }
    }
  }
}

// ---------------------------------------------------------------------------
// LSTM elementwise: gates (B x 4H, bias already fused) -> h,c,y
// ---------------------------------------------------------------------------
__device__ __forceinline__ float sigm(float x) { return 1.0f / (1.0f + expf(-x)); }

__global__ __launch_bounds__(256) void lstm_elem_kernel(
    const float* __restrict__ gates, float* __restrict__ h,
    float* __restrict__ c, float* __restrict__ y) {
  int id = blockIdx.x * blockDim.x + threadIdx.x;  // 0 .. BH-1
  int b = id >> 9, n = id & (HID - 1);
  const float* g = gates + (size_t)b * N4H;
  float ig = sigm(g[n]);
  float fg = sigm(g[HID + n]);
  float gg = tanhf(g[2 * HID + n]);
  float og = sigm(g[3 * HID + n]);
  float cn = fg * c[id] + ig * gg;
  float hn = og * tanhf(cn);
  c[id] = cn;
  h[id] = hn;
  y[id] = hn;
}

// ---------------------------------------------------------------------------
// Memory write attention (boundary steps): softmax over SLOTS of mem.wv,
// then mem = mem*(1-aw) + aw*wv.  One block per batch element, t = m index.
// ---------------------------------------------------------------------------
__global__ __launch_bounds__(256) void attn_write_kernel(
    const float* __restrict__ wv, float* __restrict__ mem) {
  __shared__ float red[256];
  __shared__ float aw[SLOTS];
  int b = blockIdx.x, t = threadIdx.x;
  float wvv = wv[(size_t)b * MSIZE + t];
  float* mb = mem + (size_t)b * SLOTS * MSIZE;
  for (int s = 0; s < SLOTS; ++s) {
    red[t] = mb[s * MSIZE + t] * wvv;
    __syncthreads();
#pragma unroll
    for (int o = 128; o > 0; o >>= 1) {
      if (t < o) red[t] += red[t + o];
      __syncthreads();
    }
    if (t == 0) aw[s] = red[0] * INV_SQRT_M;
    __syncthreads();
  }
  if (t == 0) {
    float mx = aw[0];
    for (int s = 1; s < SLOTS; ++s) mx = fmaxf(mx, aw[s]);
    float sm = 0.f;
    for (int s = 0; s < SLOTS; ++s) { aw[s] = expf(aw[s] - mx); sm += aw[s]; }
    float inv = 1.f / sm;
    for (int s = 0; s < SLOTS; ++s) aw[s] *= inv;
  }
  __syncthreads();
  for (int s = 0; s < SLOTS; ++s) {
    float a = aw[s];
    mb[s * MSIZE + t] = mb[s * MSIZE + t] * (1.f - a) + a * wvv;
  }
}

// ---------------------------------------------------------------------------
// Memory read attention: softmax over SLOTS of mem.q, read = sum ar*mem
// ---------------------------------------------------------------------------
__global__ __launch_bounds__(256) void attn_read_kernel(
    const float* __restrict__ q, const float* __restrict__ mem,
    float* __restrict__ rd) {
  __shared__ float red[256];
  __shared__ float ar[SLOTS];
  int b = blockIdx.x, t = threadIdx.x;
  float qv = q[(size_t)b * MSIZE + t];
  const float* mb = mem + (size_t)b * SLOTS * MSIZE;
  for (int s = 0; s < SLOTS; ++s) {
    red[t] = mb[s * MSIZE + t] * qv;
    __syncthreads();
#pragma unroll
    for (int o = 128; o > 0; o >>= 1) {
      if (t < o) red[t] += red[t + o];
      __syncthreads();
    }
    if (t == 0) ar[s] = red[0] * INV_SQRT_M;
    __syncthreads();
  }
  if (t == 0) {
    float mx = ar[0];
    for (int s = 1; s < SLOTS; ++s) mx = fmaxf(mx, ar[s]);
    float sm = 0.f;
    for (int s = 0; s < SLOTS; ++s) { ar[s] = expf(ar[s] - mx); sm += ar[s]; }
    float inv = 1.f / sm;
    for (int s = 0; s < SLOTS; ++s) ar[s] *= inv;
  }
  __syncthreads();
  float acc = 0.f;
  for (int s = 0; s < SLOTS; ++s) acc += ar[s] * mb[s * MSIZE + t];
  rd[(size_t)b * MSIZE + t] = acc;
}

// ---------------------------------------------------------------------------
// Slot shift register (ORDER=2): sh = [h_n, sh_old[0]]
// ---------------------------------------------------------------------------
__global__ __launch_bounds__(256) void slot_shift_kernel(
    const float* __restrict__ h, const float* __restrict__ c,
    float* __restrict__ sh, float* __restrict__ sc) {
  int id = blockIdx.x * blockDim.x + threadIdx.x;  // 0 .. BH-1
  float t0 = sh[id]; sh[BH + id] = t0; sh[id] = h[id];
  float u0 = sc[id]; sc[BH + id] = u0; sc[id] = c[id];
}

__global__ __launch_bounds__(256) void zero_f32_kernel(float* __restrict__ p, size_t n) {
  size_t id = (size_t)blockIdx.x * blockDim.x + threadIdx.x;
  if (id < n) p[id] = 0.0f;
}

__global__ __launch_bounds__(256) void copy_f32_kernel(float* __restrict__ dst,
                                                       const float* __restrict__ src,
                                                       size_t n) {
  size_t id = (size_t)blockIdx.x * blockDim.x + threadIdx.x;
  if (id < n) dst[id] = src[id];
}

// ---------------------------------------------------------------------------
extern "C" void kernel_launch(void* const* d_in, const int* in_sizes, int n_in,
                              void* d_out, int out_size, void* d_ws, size_t ws_size,
                              hipStream_t stream) {
  (void)in_sizes; (void)n_in; (void)out_size; (void)ws_size;
  const float* x  = (const float*)d_in[0];
  const float* h0 = (const float*)d_in[1];
  const float* c0 = (const float*)d_in[2];
  const float* W_ih[2] = {(const float*)d_in[3],  (const float*)d_in[10]};
  const float* W_hh[2] = {(const float*)d_in[4],  (const float*)d_in[11]};
  const float* W_mh[2] = {(const float*)d_in[5],  (const float*)d_in[12]};
  const float* bvec[2] = {(const float*)d_in[6],  (const float*)d_in[13]};
  const float* W_q[2]  = {(const float*)d_in[7],  (const float*)d_in[14]};
  const float* W_r[2]  = {(const float*)d_in[8],  (const float*)d_in[15]};
  const float* W_w[2]  = {(const float*)d_in[9],  (const float*)d_in[16]};

  // ---- workspace bump allocator (256B aligned) ----
  char* base = (char*)d_ws;
  size_t off = 0;
  auto alloc = [&](size_t bytes) -> char* {
    char* p = base + off;
    off += (bytes + 255) & ~(size_t)255;
    return p;
  };

  __bf16* WcatP[2]; __bf16* WwP[2]; __bf16* WqP[2]; __bf16* WrP[2];
  float *h[2], *c[2], *sh[2], *sc[2], *me[2], *mem[2];
  for (int l = 0; l < 2; ++l) {
    int K = (l ? HID : DIN0) + HID + 2 * HID;           // 1792 / 2048
    WcatP[l] = (__bf16*)alloc((size_t)K * N4H * 2);
    WwP[l]   = (__bf16*)alloc((size_t)(2 * HID) * MSIZE * 2);
    WqP[l]   = (__bf16*)alloc((size_t)HID * MSIZE * 2);
    WrP[l]   = (__bf16*)alloc((size_t)MSIZE * (2 * HID) * 2);
    h[l]   = (float*)alloc((size_t)BH * 4);
    c[l]   = (float*)alloc((size_t)BH * 4);
    sh[l]  = (float*)alloc((size_t)2 * BH * 4);
    sc[l]  = (float*)alloc((size_t)2 * BH * 4);
    me[l]  = (float*)alloc((size_t)BATCH * 2 * HID * 4);
    mem[l] = (float*)alloc((size_t)BATCH * SLOTS * MSIZE * 4);
  }
  __bf16* Apk   = (__bf16*)alloc((size_t)BATCH * N4H * 2);   // max M*K bf16
  float*  gates = (float*)alloc((size_t)BATCH * N4H * 4);
  float*  wv    = (float*)alloc((size_t)BATCH * MSIZE * 4);
  float*  qb    = (float*)alloc((size_t)BATCH * MSIZE * 4);
  float*  rd    = (float*)alloc((size_t)BATCH * MSIZE * 4);
  float*  y0    = (float*)alloc((size_t)BH * 4);

  // ---- prep: pack weights to bf16 fragments, init states ----
  for (int l = 0; l < 2; ++l) {
    int din = l ? HID : DIN0;
    int K = din + 3 * HID, KT = K / 32;
    pack_b_kernel<<<(K * N4H) / 256, 256, 0, stream>>>(
        W_ih[l], din, W_hh[l], HID, W_mh[l], 2 * HID, N4H, WcatP[l], KT);
    pack_b_kernel<<<(2 * HID * MSIZE) / 256, 256, 0, stream>>>(
        W_w[l], 2 * HID, nullptr, 0, nullptr, 0, MSIZE, WwP[l], (2 * HID) / 32);
    pack_b_kernel<<<(HID * MSIZE) / 256, 256, 0, stream>>>(
        W_q[l], HID, nullptr, 0, nullptr, 0, MSIZE, WqP[l], HID / 32);
    pack_b_kernel<<<(MSIZE * 2 * HID) / 256, 256, 0, stream>>>(
        W_r[l], MSIZE, nullptr, 0, nullptr, 0, 2 * HID, WrP[l], MSIZE / 32);
    copy_f32_kernel<<<BH / 256, 256, 0, stream>>>(h[l], h0, (size_t)BH);
    copy_f32_kernel<<<BH / 256, 256, 0, stream>>>(c[l], c0, (size_t)BH);
    zero_f32_kernel<<<(2 * BH) / 256, 256, 0, stream>>>(sh[l], (size_t)2 * BH);
    zero_f32_kernel<<<(2 * BH) / 256, 256, 0, stream>>>(sc[l], (size_t)2 * BH);
    zero_f32_kernel<<<(BATCH * 2 * HID) / 256, 256, 0, stream>>>(me[l], (size_t)BATCH * 2 * HID);
    zero_f32_kernel<<<(BATCH * SLOTS * MSIZE) / 256, 256, 0, stream>>>(
        mem[l], (size_t)BATCH * SLOTS * MSIZE);
  }

  float* outY = (float*)d_out;
  const int Mb = BATCH / 32;                 // 8 row-blocks of 32

  // ---- time loop, layers in lockstep ----
  for (int t = 0; t < T_STEPS; ++t) {
    for (int l = 0; l < 2; ++l) {
      int din = l ? HID : DIN0;
      int KT = (din + 3 * HID) / 32;            // 56 / 64
      const float* xt = l ? (const float*)y0 : (x + (size_t)t * BATCH * DIN0);

      // gates = [xt | h | me] @ Wcat + b   (256 x 2048)
      pack_a_kernel<<<32 * KT, 256, 0, stream>>>(
          xt, din, h[l], HID, me[l], 2 * HID, Apk, KT);
      {
        int Nb = N4H / 64;                      // 32
        wmma_gemm_kernel<<<(Mb * Nb) / 8, 256, 0, stream>>>(
            Apk, WcatP[l], bvec[l], gates, Nb, KT, N4H, Mb, /*tanh=*/0);
      }

      float* y = l ? (outY + (size_t)t * BH) : y0;
      lstm_elem_kernel<<<BH / 256, 256, 0, stream>>>(gates, h[l], c[l], y);

      if (t % 5 == 0) {
        // wv = tanh([sh0|sh1] @ W_w);   mem update;   slot shift
        pack_a_kernel<<<32 * 32, 256, 0, stream>>>(
            sh[l], HID, sh[l] + BH, HID, nullptr, 0, Apk, 32);
        wmma_gemm_kernel<<<(Mb * (MSIZE / 64)) / 8, 256, 0, stream>>>(
            Apk, WwP[l], nullptr, wv, MSIZE / 64, 32, MSIZE, Mb, /*tanh=*/1);
        attn_write_kernel<<<BATCH, 256, 0, stream>>>(wv, mem[l]);
        slot_shift_kernel<<<BH / 256, 256, 0, stream>>>(h[l], c[l], sh[l], sc[l]);
      }

      // q = h_n @ W_q ;  ar-softmax read ;  me = tanh(read @ W_r)
      pack_a_kernel<<<32 * 16, 256, 0, stream>>>(
          h[l], HID, nullptr, 0, nullptr, 0, Apk, 16);
      wmma_gemm_kernel<<<(Mb * (MSIZE / 64)) / 8, 256, 0, stream>>>(
          Apk, WqP[l], nullptr, qb, MSIZE / 64, 16, MSIZE, Mb, /*tanh=*/0);
      attn_read_kernel<<<BATCH, 256, 0, stream>>>(qb, mem[l], rd);
      pack_a_kernel<<<32 * 8, 256, 0, stream>>>(
          rd, MSIZE, nullptr, 0, nullptr, 0, Apk, 8);
      wmma_gemm_kernel<<<(Mb * (2 * HID / 64)) / 8, 256, 0, stream>>>(
          Apk, WrP[l], nullptr, me[l], (2 * HID) / 64, 8, 2 * HID, Mb, /*tanh=*/1);
    }
  }

  // ---- final (h_n, c_n) stacks ----
  float* hcout = outY + (size_t)T_STEPS * BH;
  copy_f32_kernel<<<BH / 256, 256, 0, stream>>>(hcout,            h[0], (size_t)BH);
  copy_f32_kernel<<<BH / 256, 256, 0, stream>>>(hcout + BH,       h[1], (size_t)BH);
  copy_f32_kernel<<<BH / 256, 256, 0, stream>>>(hcout + 2 * BH,   c[0], (size_t)BH);
  copy_f32_kernel<<<BH / 256, 256, 0, stream>>>(hcout + 3 * BH,   c[1], (size_t)BH);
}
